// DirGNN_75067438399963
// MI455X (gfx1250) — matrix operand order repaired and checked by reading, not compiled, not run on hardware
//
#include <hip/hip_runtime.h>
#include <math.h>

#define NN 20000
#define EE 320000
#define FF 128
#define LL 2
#define EPSV 1e-5f

typedef __bf16 bf16_t;
typedef bf16_t v16bf __attribute__((ext_vector_type(16)));
typedef float  v8f   __attribute__((ext_vector_type(8)));

// ---- WMMA k-order maps used by this file (contiguous; ISA table order differs; wave32) ----
// A-matrix 16-bit 16x32: lane L holds row M=L%16; element e -> K = hi*16 + e (this file's order)
__device__ __forceinline__ int kmapA(int e, int hi) {
    return hi * 16 + e;
}
// B-matrix 16-bit 32x16: lane L holds col N=L%16; element e -> K = hi*16 + e (this file's order)
__device__ __forceinline__ int kmapB(int e, int hi) { return hi * 16 + e; }

// monotone float<->uint encoding so uint atomicMax/Min implement float max/min
__device__ __forceinline__ unsigned fenc(float f) {
    unsigned u = __float_as_uint(f);
    return (u & 0x80000000u) ? ~u : (u | 0x80000000u);
}
__device__ __forceinline__ float fdec(unsigned u) {
    return __uint_as_float((u & 0x80000000u) ? (u & 0x7fffffffu) : ~u);
}

// ---------------- init aggregates ----------------
__global__ void k_init(float* __restrict__ s, float* __restrict__ sq,
                       unsigned* __restrict__ mne, unsigned* __restrict__ mxe,
                       float* __restrict__ cnt, float* __restrict__ avg) {
    int i = blockIdx.x * blockDim.x + threadIdx.x;
    if (i < NN * FF) { s[i] = 0.f; sq[i] = 0.f; mne[i] = 0xFFFFFFFFu; mxe[i] = 0u; }
    if (i < NN) cnt[i] = 0.f;
    if (i == 0) avg[0] = 0.f;
}

// ---------------- in-degree ----------------
__global__ void k_degree(const int* __restrict__ dst, float* __restrict__ cnt) {
    int i = blockIdx.x * blockDim.x + threadIdx.x;
    if (i < EE) atomicAdd(&cnt[dst[i]], 1.0f);
}

// ---------------- sum(log1p(deg)) ----------------
__global__ void k_avglog(const float* __restrict__ cnt, float* __restrict__ avg) {
    int i = blockIdx.x * blockDim.x + threadIdx.x;
    if (i < NN) atomicAdd(avg, log1pf(cnt[i]));
}

// ---- pack fp32 weight W[128,K] (out = h @ W.T) into per-lane WMMA B layout ----
__global__ void k_pack(const float* __restrict__ W, bf16_t* __restrict__ P, int K) {
    int idx = blockIdx.x * blockDim.x + threadIdx.x;
    if (idx >= 128 * K) return;
    int e   = idx & 15;
    int L   = (idx >> 4) & 31;
    int kst = K >> 5;
    int ks  = (idx >> 9) % kst;
    int ct  = (idx >> 9) / kst;
    int n   = ct * 16 + (L & 15);
    int k   = ks * 32 + kmapB(e, L >> 4);
    P[idx]  = (bf16_t)W[n * K + k];
}

// ---- fused message GEMM + multi-aggregator scatter -------------------------
// tile = 16 edges; A = bf16(cat(x[dst], x[src])) [16 x 256] staged in LDS;
// B = packed pre_w; 4 waves x 2 col-tiles cover all 128 output features.
__global__ void __launch_bounds__(128) k_message(
    const float* __restrict__ x, const int* __restrict__ src, const int* __restrict__ dst,
    const bf16_t* __restrict__ Wp, const float* __restrict__ bias,
    float* __restrict__ s, float* __restrict__ sq,
    unsigned* __restrict__ mne, unsigned* __restrict__ mxe)
{
    __shared__ bf16_t hbuf[16 * 256];
    __shared__ int dsts[16];
    const int t  = threadIdx.x;
    const int e0 = blockIdx.x * 16;
    if (t < 16) dsts[t] = dst[e0 + t];
    for (int i = t; i < 16 * 256; i += 128) {
        int e = i >> 8, c = i & 255;
        int node = (c < 128) ? dst[e0 + e] : src[e0 + e];
        hbuf[i] = (bf16_t)x[node * FF + (c & 127)];
    }
    __syncthreads();

    const int w = t >> 5, L = t & 31, r = L & 15, hi = L >> 4;
    v8f acc0 = {}; v8f acc1 = {};
    for (int ks = 0; ks < 8; ++ks) {
        v16bf a;
#pragma unroll
        for (int e = 0; e < 16; ++e) a[e] = hbuf[r * 256 + ks * 32 + kmapA(e, hi)];
        const v16bf b0 = *(const v16bf*)(Wp + (size_t)(((w    ) * 8 + ks) * 32 + L) * 16);
        const v16bf b1 = *(const v16bf*)(Wp + (size_t)(((w + 4) * 8 + ks) * 32 + L) * 16);
        acc0 = __builtin_amdgcn_wmma_f32_16x16x32_bf16(false, a, false, b0, (short)0, acc0, false, false);
        acc1 = __builtin_amdgcn_wmma_f32_16x16x32_bf16(false, a, false, b1, (short)0, acc1, false, false);
    }
#pragma unroll
    for (int i = 0; i < 8; ++i) {
        int m = hi * 8 + i;                 // edge row in tile (C/D layout)
        int dnode = dsts[m];
        {
            int col = w * 16 + r;
            float v = acc0[i] + bias[col];
            size_t o = (size_t)dnode * FF + col;
            atomicAdd(&s[o], v); atomicAdd(&sq[o], v * v);
            atomicMax(&mxe[o], fenc(v)); atomicMin(&mne[o], fenc(v));
        }
        {
            int col = (w + 4) * 16 + r;
            float v = acc1[i] + bias[col];
            size_t o = (size_t)dnode * FF + col;
            atomicAdd(&s[o], v); atomicAdd(&sq[o], v * v);
            atomicMax(&mxe[o], fenc(v)); atomicMin(&mne[o], fenc(v));
        }
    }
}

// ---- build [N, 16F] bf16 PNA feature matrix: [x | agg | agg*s1 | agg*s2] ----
__global__ void k_build16(const float* __restrict__ x, const float* __restrict__ s,
                          const float* __restrict__ sq, const unsigned* __restrict__ mne,
                          const unsigned* __restrict__ mxe, const float* __restrict__ cnt,
                          const float* __restrict__ avgsum, bf16_t* __restrict__ A)
{
    int i = blockIdx.x * blockDim.x + threadIdx.x;
    if (i >= NN * FF) return;
    int n = i >> 7, f = i & 127;
    float c    = cnt[n];
    float degc = fmaxf(c, 1.f);
    float sv   = s[i];
    float mean = sv / degc;
    float var  = sq[i] / degc - mean * mean;
    float sd   = sqrtf(fmaxf(var, 0.f) + EPSV);
    float mnv  = (c > 0.f) ? fdec(mne[i]) : 0.f;
    float mxv  = (c > 0.f) ? fdec(mxe[i]) : 0.f;
    float avg  = avgsum[0] / (float)NN;
    float logd = logf(degc + 1.f);
    float s1 = logd / avg, s2 = avg / logd;
    bf16_t* row = A + (size_t)n * 2048;
    row[f]        = (bf16_t)x[i];
    row[128 + f]  = (bf16_t)mean;       row[256 + f]  = (bf16_t)sv;
    row[384 + f]  = (bf16_t)sd;         row[512 + f]  = (bf16_t)mnv;
    row[640 + f]  = (bf16_t)mxv;
    row[768 + f]  = (bf16_t)(mean * s1); row[896 + f]  = (bf16_t)(sv * s1);
    row[1024 + f] = (bf16_t)(sd * s1);   row[1152 + f] = (bf16_t)(mnv * s1);
    row[1280 + f] = (bf16_t)(mxv * s1);
    row[1408 + f] = (bf16_t)(mean * s2); row[1536 + f] = (bf16_t)(sv * s2);
    row[1664 + f] = (bf16_t)(sd * s2);   row[1792 + f] = (bf16_t)(mnv * s2);
    row[1920 + f] = (bf16_t)(mxv * s2);
}

// ---- generic WMMA GEMM: C[M,128] = A_bf16[M,K] @ Wp + bias, opt relu ----
__global__ void __launch_bounds__(128) k_gemm(
    const bf16_t* __restrict__ A, int K, const bf16_t* __restrict__ Wp,
    const float* __restrict__ bias, float* __restrict__ outF,
    bf16_t* __restrict__ outB, int relu)
{
    extern __shared__ bf16_t lds[];
    const int t = threadIdx.x;
    // stage 16 rows of A into LDS (dword copies, coalesced)
    const unsigned* g = (const unsigned*)(A + (size_t)blockIdx.x * 16 * K);
    unsigned* l = (unsigned*)lds;
    const int ndw = (16 * K) >> 1;
    for (int i = t; i < ndw; i += 128) l[i] = g[i];
    __syncthreads();

    const int w = t >> 5, L = t & 31, r = L & 15, hi = L >> 4;
    const int kst = K >> 5;
    v8f acc0 = {}; v8f acc1 = {};
    for (int ks = 0; ks < kst; ++ks) {
        v16bf a;
#pragma unroll
        for (int e = 0; e < 16; ++e) a[e] = lds[r * K + ks * 32 + kmapA(e, hi)];
        const v16bf b0 = *(const v16bf*)(Wp + (size_t)(((w    ) * kst + ks) * 32 + L) * 16);
        const v16bf b1 = *(const v16bf*)(Wp + (size_t)(((w + 4) * kst + ks) * 32 + L) * 16);
        acc0 = __builtin_amdgcn_wmma_f32_16x16x32_bf16(false, a, false, b0, (short)0, acc0, false, false);
        acc1 = __builtin_amdgcn_wmma_f32_16x16x32_bf16(false, a, false, b1, (short)0, acc1, false, false);
    }
#pragma unroll
    for (int i = 0; i < 8; ++i) {
        int m = hi * 8 + i;
        size_t row = (size_t)(blockIdx.x * 16 + m) * 128;
        {
            int col = w * 16 + r;
            float v = acc0[i] + bias[col];
            if (relu) v = fmaxf(v, 0.f);
            if (outF) outF[row + col] = v;
            if (outB) outB[row + col] = (bf16_t)v;
        }
        {
            int col = (w + 4) * 16 + r;
            float v = acc1[i] + bias[col];
            if (relu) v = fmaxf(v, 0.f);
            if (outF) outF[row + col] = v;
            if (outB) outB[row + col] = (bf16_t)v;
        }
    }
}

// ---- build [N, 3F] bf16 = cat(x, h_in, h_out) ----
__global__ void k_combin(const float* __restrict__ x, const float* __restrict__ hin,
                         const float* __restrict__ hout, bf16_t* __restrict__ A) {
    int i = blockIdx.x * blockDim.x + threadIdx.x;
    if (i >= NN * 384) return;
    int n = i / 384, c = i % 384;
    float v = (c < 128) ? x[n * 128 + c]
            : (c < 256) ? hin[n * 128 + c - 128]
                        : hout[n * 128 + c - 256];
    A[(size_t)n * 384 + c] = (bf16_t)v;
}

// ---- final projection to scalar: one wave32 per node ----
__global__ void k_out(const float* __restrict__ x, const float* __restrict__ ow,
                      const float* __restrict__ ob, float* __restrict__ y) {
    int wid  = (blockIdx.x * blockDim.x + threadIdx.x) >> 5;
    int lane = threadIdx.x & 31;
    if (wid >= NN) return;
    float acc = 0.f;
    for (int f = lane; f < 128; f += 32) acc += x[(size_t)wid * 128 + f] * ow[f];
    for (int off = 16; off; off >>= 1) acc += __shfl_down(acc, off, 32);
    if (lane == 0) y[wid] = acc + ob[0];
}

extern "C" void kernel_launch(void* const* d_in, const int* in_sizes, int n_in,
                              void* d_out, int out_size, void* d_ws, size_t ws_size,
                              hipStream_t stream)
{
    (void)in_sizes; (void)n_in; (void)out_size; (void)ws_size;
    const float* x0     = (const float*)d_in[0];
    const int*   ein    = (const int*)d_in[1];
    const int*   eout   = (const int*)d_in[2];
    const float* pre_w  = (const float*)d_in[3];
    const float* pre_b  = (const float*)d_in[4];
    const float* post_w = (const float*)d_in[5];
    const float* post_b = (const float*)d_in[6];
    const float* lin_w  = (const float*)d_in[7];
    const float* lin_b  = (const float*)d_in[8];
    const float* comb_w = (const float*)d_in[9];
    const float* comb_b = (const float*)d_in[10];
    const float* out_w  = (const float*)d_in[11];
    const float* out_b  = (const float*)d_in[12];
    float* y = (float*)d_out;

    char* ws = (char*)d_ws;
    size_t off = 0;
    auto alloc = [&](size_t bytes) -> void* {
        void* p = ws + off; off += (bytes + 255) & ~(size_t)255; return p;
    };
    float*    x_cur = (float*)alloc((size_t)NN * FF * 4);
    float*    h_in  = (float*)alloc((size_t)NN * FF * 4);
    float*    h_out = (float*)alloc((size_t)NN * FF * 4);
    float*    sbuf  = (float*)alloc((size_t)NN * FF * 4);
    float*    sqbuf = (float*)alloc((size_t)NN * FF * 4);
    unsigned* mne   = (unsigned*)alloc((size_t)NN * FF * 4);
    unsigned* mxe   = (unsigned*)alloc((size_t)NN * FF * 4);
    float*    cnt   = (float*)alloc((size_t)NN * 4);
    float*    avg   = (float*)alloc(256);
    bf16_t*   Abuf  = (bf16_t*)alloc((size_t)NN * 2048 * 2);
    bf16_t*   Tbuf  = (bf16_t*)alloc((size_t)NN * FF * 2);
    bf16_t*   Wpack = (bf16_t*)alloc((size_t)128 * 2048 * 2);

    hipMemcpyAsync(x_cur, x0, (size_t)NN * FF * 4, hipMemcpyDeviceToDevice, stream);

    for (int l = 0; l < LL; ++l) {
        for (int d = 0; d < 2; ++d) {
            const int* src  = (d == 0) ? ein : eout;
            const int* dstp = src + EE;
            const float* pw = pre_w  + (size_t)(l * 2 + d) * FF * 2 * FF;
            const float* pb = pre_b  + (size_t)(l * 2 + d) * FF;
            const float* qw = post_w + (size_t)(l * 2 + d) * FF * 16 * FF;
            const float* qb = post_b + (size_t)(l * 2 + d) * FF;
            const float* lw = lin_w  + (size_t)(l * 2 + d) * FF * FF;
            const float* lb = lin_b  + (size_t)(l * 2 + d) * FF;
            float* hdir = (d == 0) ? h_in : h_out;

            k_init<<<(NN * FF + 255) / 256, 256, 0, stream>>>(sbuf, sqbuf, mne, mxe, cnt, avg);
            k_degree<<<(EE + 255) / 256, 256, 0, stream>>>(dstp, cnt);
            k_pack<<<(128 * 256 + 255) / 256, 256, 0, stream>>>(pw, Wpack, 256);
            k_message<<<EE / 16, 128, 0, stream>>>(x_cur, src, dstp, Wpack, pb,
                                                   sbuf, sqbuf, mne, mxe);
            k_avglog<<<(NN + 255) / 256, 256, 0, stream>>>(cnt, avg);
            k_build16<<<(NN * FF + 255) / 256, 256, 0, stream>>>(x_cur, sbuf, sqbuf,
                                                                 mne, mxe, cnt, avg, Abuf);
            k_pack<<<(128 * 2048 + 255) / 256, 256, 0, stream>>>(qw, Wpack, 2048);
            k_gemm<<<NN / 16, 128, 16 * 2048 * 2, stream>>>(Abuf, 2048, Wpack, qb,
                                                            nullptr, Tbuf, 0);
            k_pack<<<(128 * 128 + 255) / 256, 256, 0, stream>>>(lw, Wpack, 128);
            k_gemm<<<NN / 16, 128, 16 * 128 * 2, stream>>>(Tbuf, 128, Wpack, lb,
                                                           hdir, nullptr, 1);
        }
        k_combin<<<(NN * 384 + 255) / 256, 256, 0, stream>>>(x_cur, h_in, h_out, Abuf);
        k_pack<<<(128 * 384 + 255) / 256, 256, 0, stream>>>(comb_w + (size_t)l * FF * 3 * FF,
                                                            Wpack, 384);
        k_gemm<<<NN / 16, 128, 16 * 384 * 2, stream>>>(Abuf, 384, Wpack,
                                                       comb_b + (size_t)l * FF,
                                                       x_cur, nullptr, 1);
    }
    k_out<<<(NN * 32 + 127) / 128, 128, 0, stream>>>(x_cur, out_w, out_b, y);
}
